// EnhancedGovernanceAttention_80522046865761
// MI455X (gfx1250) — matrix operand order repaired and checked by reading, not compiled, not run on hardware
//
#include <hip/hip_runtime.h>
#include <hip/hip_bf16.h>

// ---------------------------------------------------------------------------
// EnhancedGovernanceAttention for MI455X (gfx1250, wave32, WMMA)
//
// Pipeline: cvt(x,W*)->f16  |  fused QKV GEMM + in-register RoPE (WMMA f16)
//           | flash attention w/ fused mask-bias (WMMA f16, LDS P-reshape)
//           | out-projection GEMM + bias (WMMA f16, fp32 out)
// ---------------------------------------------------------------------------

#define HIDDEN   2048
#define HEADS    16
#define HEAD_DIM 128
#define SEQ      1024
#define BATCH    2
#define NTOK     (BATCH * SEQ)

typedef _Float16 h16;
typedef __attribute__((ext_vector_type(16))) _Float16 v16h;
typedef __attribute__((ext_vector_type(8)))  _Float16 v8h;
typedef __attribute__((ext_vector_type(8)))  float    v8f;

// ---- WMMA fragment helpers (16-bit, k = 16h+s in A and B alike) ----------
// A 16x32 (MxK) row-major source: lane r (0..15) holds row r; half h=lane/16
// covers K = [16h .. 16h+15] (contiguous per lane).
__device__ __forceinline__ v16h frag_a_rowmajor(const h16* base, int ld) {
    const int lane = threadIdx.x & 31;
    const int hf = lane >> 4, r = lane & 15;
    const h16* p = base + r * ld + hf * 16;
    union { v16h v; v8h h[2]; } f;
    f.h[0] = *(const v8h*)(p);
    f.h[1] = *(const v8h*)(p + 8);
    return f.v;
}

// B 32x16 (KxN) from "K-contiguous" storage srcT[n*ldk + k]: lane c = column,
// group g = lane/16 covers K = [16g .. 16g+15] (contiguous per lane).
__device__ __forceinline__ v16h frag_b_kcontig(const h16* baseT, int ldk) {
    const int lane = threadIdx.x & 31;
    const int g = lane >> 4, c = lane & 15;
    const h16* p = baseT + c * ldk + g * 16;
    union { v16h v; v8h h[2]; } f;
    f.h[0] = *(const v8h*)(p);
    f.h[1] = *(const v8h*)(p + 8);
    return f.v;
}

__device__ __forceinline__ v8f wmma16(v16h a, v16h b, v8f c) {
    // v_wmma_f32_16x16x32_f16 (probe-verified builtin, 8 args)
    return __builtin_amdgcn_wmma_f32_16x16x32_f16(
        false, a, false, b, (short)0, c, false, false);
}

// ---- fp32 -> f16 convert --------------------------------------------------
__global__ void k_cvt(const float* __restrict__ s, h16* __restrict__ d, int n) {
    const int i = blockIdx.x * blockDim.x + threadIdx.x;
    if (i < n) d[i] = (h16)s[i];
}

// ---- fp32 KxN -> f16 NxK (transpose) via 32x33 LDS tile -------------------
__global__ void k_cvt_t(const float* __restrict__ s, h16* __restrict__ d) {
    __shared__ float tile[32][33];
    const int tx = threadIdx.x & 31, ty = threadIdx.x >> 5;   // 256 thr: ty 0..7
    const int kt = blockIdx.x * 32, nt = blockIdx.y * 32;
    #pragma unroll
    for (int j = 0; j < 32; j += 8)
        tile[ty + j][tx] = s[(size_t)(kt + ty + j) * HIDDEN + nt + tx];
    __syncthreads();
    #pragma unroll
    for (int j = 0; j < 32; j += 8)
        d[(size_t)(nt + ty + j) * HIDDEN + kt + tx] = (h16)tile[tx][ty + j];
}

// ---- GEMM: modes 0=Q(+RoPE) 1=K(+RoPE) 2=V(store transposed) 3=out-proj ---
__global__ void __launch_bounds__(256)
k_gemm(const h16* __restrict__ A,
       const h16* __restrict__ BTq, const h16* __restrict__ BTk,
       const h16* __restrict__ BTv,
       h16* __restrict__ dq, h16* __restrict__ dk, h16* __restrict__ dvT,
       float* __restrict__ outf, const float* __restrict__ bias, int modeParam)
{
    const int mode = (modeParam < 0) ? (int)blockIdx.z : modeParam;
    const h16* BT = (mode == 1) ? BTk : (mode == 2) ? BTv : BTq;

    const int wave = threadIdx.x >> 5;
    const int lane = threadIdx.x & 31;
    const int hf = lane >> 4, col = lane & 15;
    const int row0 = blockIdx.x * 128 + wave * 16;   // token rows (16 per wave)
    const int n0   = blockIdx.y * 128;               // output columns

    const h16* Aw = A + (size_t)row0 * HIDDEN;
    const h16* Bt = BT + (size_t)n0 * HIDDEN;

    const v8f vzero = {0.f,0.f,0.f,0.f,0.f,0.f,0.f,0.f};
    v8f acc[8];
    #pragma unroll
    for (int t = 0; t < 8; ++t) acc[t] = vzero;

    for (int kk = 0; kk < HIDDEN; kk += 32) {
        const v16h a = frag_a_rowmajor(Aw + kk, HIDDEN);
        #pragma unroll
        for (int t = 0; t < 8; ++t) {
            const v16h b = frag_b_kcontig(Bt + (size_t)(t * 16) * HIDDEN + kk, HIDDEN);
            acc[t] = wmma16(a, b, acc[t]);
        }
    }

    if (mode == 3) {                                 // out = acc + bo, fp32
        #pragma unroll
        for (int t = 0; t < 8; ++t) {
            const int n = n0 + t * 16 + col;
            const float bb = bias[n];
            #pragma unroll
            for (int i = 0; i < 8; ++i)
                outf[(size_t)(row0 + i + 8 * hf) * HIDDEN + n] = acc[t][i] + bb;
        }
        return;
    }

    const int b    = row0 >> 10;                     // batch (S=1024)
    const int s0   = row0 & (SEQ - 1);
    const int head = blockIdx.y;                     // 128-wide tile == 1 head
    const int bh   = b * HEADS + head;

    if (mode == 2) {                                 // V -> vT[bh][d][s]
        #pragma unroll
        for (int t = 0; t < 8; ++t) {
            const int d = t * 16 + col;
            v8h pack;
            #pragma unroll
            for (int i = 0; i < 8; ++i) pack[i] = (h16)acc[t][i];
            *(v8h*)(dvT + ((size_t)bh * HEAD_DIM + d) * SEQ + s0 + 8 * hf) = pack;
        }
        return;
    }

    // RoPE in-register: pair (d, d+64) lives in acc[t][i] / acc[t+4][i]
    #pragma unroll
    for (int t = 0; t < 4; ++t) {
        const int j = t * 16 + col;                  // freq index 0..63
        const float invf = exp2f(-(float)j * (13.287712379549449f / 64.0f));
        #pragma unroll
        for (int i = 0; i < 8; ++i) {
            const int s = s0 + i + 8 * hf;
            float sn, cs;
            __sincosf((float)s * invf, &sn, &cs);
            const float x1 = acc[t][i], x2 = acc[t + 4][i];
            acc[t][i]     = x1 * cs - x2 * sn;
            acc[t + 4][i] = x2 * cs + x1 * sn;
        }
    }
    h16* dst = (mode == 0) ? dq : dk;                // [bh][s][d]
    #pragma unroll
    for (int t = 0; t < 8; ++t) {
        const int d = t * 16 + col;
        #pragma unroll
        for (int i = 0; i < 8; ++i)
            dst[((size_t)bh * SEQ + s0 + i + 8 * hf) * HEAD_DIM + d] = (h16)acc[t][i];
    }
}

// ---- Flash attention with fused governance bias ---------------------------
__global__ void __launch_bounds__(256)
k_attn(const h16* __restrict__ qg, const h16* __restrict__ kg,
       const h16* __restrict__ vT,
       const float* __restrict__ pm, const float* __restrict__ pol,
       const float* __restrict__ mw, h16* __restrict__ ctx)
{
    __shared__ __align__(16) h16 smem[8][16 * 32];   // per-wave P staging
    const int bh   = blockIdx.y;
    const int wave = threadIdx.x >> 5;
    const int lane = threadIdx.x & 31;
    const int hf = lane >> 4, col = lane & 15;
    const int q0 = blockIdx.x * 128 + wave * 16;
    const float scale = 0.08838834764831845f;        // 1/sqrt(128)
    const float NEGINF = -__builtin_inff();

    const h16* qrow = qg + ((size_t)bh * SEQ + q0) * HEAD_DIM;
    v16h aq[4];
    #pragma unroll
    for (int t = 0; t < 4; ++t) aq[t] = frag_a_rowmajor(qrow + t * 32, HEAD_DIM);

    const v8f vzero = {0.f,0.f,0.f,0.f,0.f,0.f,0.f,0.f};
    v8f o[8];
    #pragma unroll
    for (int t = 0; t < 8; ++t) o[t] = vzero;
    float mrun[8], lrun[8];
    #pragma unroll
    for (int i = 0; i < 8; ++i) { mrun[i] = NEGINF; lrun[i] = 0.f; }

    h16* pl = &smem[wave][0];
    const int kbmax = (q0 + 15) >> 5;                // causal key-block bound

    for (int kb = 0; kb <= kbmax; ++kb) {
        const int k0 = kb * 32;

        if (kb < kbmax) {                            // prefetch next mask block
            const size_t poff = ((size_t)bh * SEQ + q0 + col) * SEQ + k0 + 32;
            __builtin_prefetch(pm + poff, 0, 0);
            __builtin_prefetch(pol + poff, 0, 0);
            __builtin_prefetch(mw + poff, 0, 0);
        }

        // ---- scores: Q(16x128) x K^T(128x32) + bias, causal mask ----
        v8f sc[2];
        #pragma unroll
        for (int sub = 0; sub < 2; ++sub) {
            v8f s = vzero;
            const h16* kbase = kg + ((size_t)bh * SEQ + k0 + sub * 16) * HEAD_DIM;
            #pragma unroll
            for (int t = 0; t < 4; ++t) {
                const v16h bk = frag_b_kcontig(kbase + t * 32, HEAD_DIM);
                s = wmma16(aq[t], bk, s);
            }
            const int kgi = k0 + sub * 16 + col;
            #pragma unroll
            for (int i = 0; i < 8; ++i) {
                const int qi = q0 + i + 8 * hf;
                if (kgi <= qi) {
                    const size_t off = ((size_t)bh * SEQ + qi) * SEQ + kgi;
                    const float bias = pm[off] * 0.05f + pol[off] * 0.1f
                                     + __logf(1.0f + mw[off] * 0.1f + 1e-8f);
                    s[i] = s[i] * scale + bias;
                } else {
                    s[i] = NEGINF;
                }
            }
            sc[sub] = s;
        }

        // ---- online softmax (16-lane row reductions) ----
        float rm[8];
        #pragma unroll
        for (int i = 0; i < 8; ++i) rm[i] = fmaxf(sc[0][i], sc[1][i]);
        #pragma unroll
        for (int off = 8; off > 0; off >>= 1)
            #pragma unroll
            for (int i = 0; i < 8; ++i)
                rm[i] = fmaxf(rm[i], __shfl_xor(rm[i], off, 16));

        float corr[8];
        #pragma unroll
        for (int i = 0; i < 8; ++i) {
            const float nm = fmaxf(mrun[i], rm[i]);
            corr[i] = __expf(mrun[i] - nm);
            mrun[i] = nm;
        }

        float ps[8];
        #pragma unroll
        for (int i = 0; i < 8; ++i) ps[i] = 0.f;
        #pragma unroll
        for (int sub = 0; sub < 2; ++sub)
            #pragma unroll
            for (int i = 0; i < 8; ++i) {
                const float p = __expf(sc[sub][i] - mrun[i]);  // -inf -> 0
                ps[i] += p;
                pl[(i + 8 * hf) * 32 + sub * 16 + col] = (h16)p;
            }
        #pragma unroll
        for (int off = 8; off > 0; off >>= 1)
            #pragma unroll
            for (int i = 0; i < 8; ++i)
                ps[i] += __shfl_xor(ps[i], off, 16);
        #pragma unroll
        for (int i = 0; i < 8; ++i) lrun[i] = lrun[i] * corr[i] + ps[i];
        #pragma unroll
        for (int t = 0; t < 8; ++t)
            #pragma unroll
            for (int i = 0; i < 8; ++i) o[t][i] *= corr[i];

        // P written cross-lane to LDS; drain DScnt before fragment reload
        asm volatile("s_wait_dscnt 0x0" ::: "memory");

        // ---- O += P(16x32) x V(32x128) ----
        const v16h ap = frag_a_rowmajor(pl, 32);
        #pragma unroll
        for (int t = 0; t < 8; ++t) {
            const v16h bv = frag_b_kcontig(
                vT + ((size_t)bh * HEAD_DIM + t * 16) * SEQ + k0, SEQ);
            o[t] = wmma16(ap, bv, o[t]);
        }
    }

    // ---- normalize + store ctx as [b, s, h*Dh] f16 ----
    const int b = bh >> 4, head = bh & 15;
    float linv[8];
    #pragma unroll
    for (int i = 0; i < 8; ++i) linv[i] = 1.0f / lrun[i];
    #pragma unroll
    for (int t = 0; t < 8; ++t) {
        const int d = head * HEAD_DIM + t * 16 + col;
        #pragma unroll
        for (int i = 0; i < 8; ++i)
            ctx[((size_t)b * SEQ + q0 + i + 8 * hf) * HIDDEN + d] =
                (h16)(o[t][i] * linv[i]);
    }
}

// ---------------------------------------------------------------------------
extern "C" void kernel_launch(void* const* d_in, const int* in_sizes, int n_in,
                              void* d_out, int out_size, void* d_ws, size_t ws_size,
                              hipStream_t stream) {
    const float* x   = (const float*)d_in[0];
    const float* pm  = (const float*)d_in[1];
    const float* pol = (const float*)d_in[2];
    const float* mw  = (const float*)d_in[3];
    const float* Wq  = (const float*)d_in[4];
    const float* Wk  = (const float*)d_in[5];
    const float* Wv  = (const float*)d_in[6];
    const float* Wo  = (const float*)d_in[7];
    const float* bo  = (const float*)d_in[8];
    float* out = (float*)d_out;

    const size_t E = (size_t)HIDDEN * HIDDEN;        // 4 Mi elements
    if (ws_size < 9 * E * sizeof(h16)) return;       // need 72 MiB scratch
    h16* ws  = (h16*)d_ws;
    h16* xb  = ws;                                   // x in f16
    h16* WqT = ws + 1 * E;                           // weights, N x K f16
    h16* WkT = ws + 2 * E;
    h16* WvT = ws + 3 * E;
    h16* WoT = ws + 4 * E;
    h16* qb  = ws + 5 * E;                           // roped Q [bh][s][d]
    h16* kbuf= ws + 6 * E;                           // roped K [bh][s][d]
    h16* vT  = ws + 7 * E;                           // V transposed [bh][d][s]
    h16* ctx = ws + 8 * E;                           // attention context

    k_cvt<<<dim3((unsigned)(E / 256)), 256, 0, stream>>>(x, xb, (int)E);
    const dim3 tg(HIDDEN / 32, HIDDEN / 32);
    k_cvt_t<<<tg, 256, 0, stream>>>(Wq, WqT);
    k_cvt_t<<<tg, 256, 0, stream>>>(Wk, WkT);
    k_cvt_t<<<tg, 256, 0, stream>>>(Wv, WvT);
    k_cvt_t<<<tg, 256, 0, stream>>>(Wo, WoT);

    k_gemm<<<dim3(NTOK / 128, HIDDEN / 128, 3), 256, 0, stream>>>(
        xb, WqT, WkT, WvT, qb, kbuf, vT, nullptr, nullptr, -1);

    k_attn<<<dim3(SEQ / 128, BATCH * HEADS), 256, 0, stream>>>(
        qb, kbuf, vT, pm, pol, mw, ctx);

    k_gemm<<<dim3(NTOK / 128, HIDDEN / 128, 1), 256, 0, stream>>>(
        ctx, WoT, WoT, WoT, nullptr, nullptr, nullptr, out, bo, 3);
}
